// HierarchicalBinder_38843684225995
// MI455X (gfx1250) — matrix operand order repaired and checked by reading, not compiled, not run on hardware
//
#include <hip/hip_runtime.h>
#include <math.h>

#define B_  8
#define T_  16384
#define D_  128
#define BT_ (B_ * T_)

typedef __attribute__((ext_vector_type(16))) __bf16 v16bf;
typedef __attribute__((ext_vector_type(8)))  float  v8f;

static __device__ __forceinline__ unsigned short f32_bf16(float f) {
  union { float f; unsigned u; } c; c.f = f;
  unsigned r = c.u + 0x7FFFu + ((c.u >> 16) & 1u);   // round-to-nearest-even
  return (unsigned short)(r >> 16);
}

// ---- CDNA5 async global->LDS (ASYNCcnt) -----------------------------------
// LDS destination address comes from the VDST VGPR plus offset; generic LDS pointer low 32 bits
// are the in-aperture byte offset (flat LDS mapping truncates to addr[31:0]).
static __device__ __forceinline__ unsigned lds_off(const void* p) {
  return (unsigned)(unsigned long long)p;
}
static __device__ __forceinline__ void g2l_async_b128(unsigned lds, unsigned long long ga) {
  asm volatile("global_load_async_to_lds_b128 %0, %1, off" :: "v"(lds), "v"(ga) : "memory");
}
static __device__ __forceinline__ void wait_async0() {
  asm volatile("s_wait_asynccnt 0x0" ::: "memory");
}
// copy `bytes` (multiple of 2048) from 16B-aligned global to 16B-aligned LDS, 128 threads
static __device__ __forceinline__ void stage_async(const void* lds, const void* g, int bytes) {
  const unsigned l0 = lds_off(lds) + threadIdx.x * 16;
  const unsigned long long g0 = (unsigned long long)g + threadIdx.x * 16;
  for (int o = 0; o < bytes; o += 2048) g2l_async_b128(l0 + o, g0 + o);
}

// Build a v16bf WMMA operand from two 16-byte contiguous LDS runs (2x ds_load_b128).
static __device__ __forceinline__ v16bf frag_from2(const unsigned short* p0,
                                                   const unsigned short* p1) {
  union { uint4 q[2]; v16bf v; } u;
  u.q[0] = *(const uint4*)p0;
  u.q[1] = *(const uint4*)p1;
  return u.v;
}

// ---- WMMA fragment helpers (ISA 7.12.2 layouts, wave32) -------------------
// A fragment: 16x32 bf16 from row-major tile (stride S shorts).
static __device__ __forceinline__ v16bf frag_a(const unsigned short* t, int row0, int k0, int S) {
  const int lane = threadIdx.x & 31;
  const unsigned short* p = t + (row0 + (lane & 15)) * S + k0 + ((lane >> 4) << 3);
  return frag_from2(p, p + 16);
}
// B fragment: 32x16 bf16 from COLUMN-MAJOR weight tile t[n*SK + k] (k contiguous).
static __device__ __forceinline__ v16bf frag_b(const unsigned short* t, int k0, int col0, int SK) {
  const int lane = threadIdx.x & 31;
  const unsigned short* p = t + (col0 + (lane & 15)) * SK + k0 + ((lane >> 4) << 3);
  return frag_from2(p, p + 16);
}
// C fragment store: lane L -> column col0+(L&15), rows row0+(L>>4)*8 + v.
static __device__ __forceinline__ void store_c(float* dst, int S, int row0, int col0,
                                               v8f c, const float* bias) {
  const int lane = threadIdx.x & 31;
  const int col  = col0 + (lane & 15);
  float* p = dst + (size_t)(row0 + ((lane >> 4) << 3)) * S + col;
  const float bb = bias[col];
#pragma unroll
  for (int v = 0; v < 8; ++v) p[(size_t)v * S] = c[v] + bb;
}

// ---- Kernel 0: one-time prep: fold Wcomb, emit bf16 col-major weight images
__global__ void k_prep(const float* __restrict__ Wk,  const float* __restrict__ Wq,
                       const float* __restrict__ Wa1,
                       const float* __restrict__ Wa2, const float* __restrict__ ba2,
                       const float* __restrict__ Wd,  const float* __restrict__ bd,
                       unsigned short* __restrict__ uWkT, unsigned short* __restrict__ uWqT,
                       unsigned short* __restrict__ uW1T, unsigned short* __restrict__ uWcT,
                       float* __restrict__ bc) {
  const int idx = blockIdx.x * 256 + threadIdx.x;       // 16384 threads
  if (idx < 128 * 64) {                                 // [k][n] -> colmajor [n][k]
    const int k = idx >> 6, n = idx & 63;
    uWkT[n * 128 + k] = f32_bf16(Wk[idx]);
    uWqT[n * 128 + k] = f32_bf16(Wq[idx]);
  }
  {                                                     // Wa1 and Wcomb = Wa2@Wd
    const int k = idx >> 7, n = idx & 127;
    uW1T[n * 128 + k] = f32_bf16(Wa1[idx]);
    float s = 0.f;
    for (int j = 0; j < 128; ++j) s += Wa2[k * 128 + j] * Wd[j * 128 + n];
    uWcT[n * 128 + k] = f32_bf16(s);
  }
  if (idx < 128) {                                      // bcomb = ba2@Wd + bd
    float s = bd[idx];
    for (int j = 0; j < 128; ++j) s += ba2[j] * Wd[j * 128 + idx];
    bc[idx] = s;
  }
}

// ---- Kernel 1: K = xWk + bk, Q = xWq + bq  (bf16 WMMA, async weight stage) -
__global__ __launch_bounds__(128) void k_proj(const float* __restrict__ x,
    const unsigned short* __restrict__ uWkT, const unsigned short* __restrict__ uWqT,
    const float* __restrict__ bk, const float* __restrict__ bq,
    float* __restrict__ Kb, float* __restrict__ Qb) {
  __shared__ __align__(16) unsigned short sX[64 * 128];   // row-major (A side)
  __shared__ __align__(16) unsigned short sK[64 * 128];   // col-major [n][k]
  __shared__ __align__(16) unsigned short sQ[64 * 128];   // col-major [n][k]
  __shared__ float sbk[64], sbq[64];
  const int tid = threadIdx.x;
  const long r0 = (long)blockIdx.x * 64;
  stage_async(sK, uWkT, 64 * 128 * 2);                    // DMA, no VGPR staging
  stage_async(sQ, uWqT, 64 * 128 * 2);
  for (int i = tid; i < 64 * 128; i += 128) sX[i] = f32_bf16(x[r0 * 128 + i]);
  if (tid < 64) { sbk[tid] = bk[tid]; sbq[tid] = bq[tid]; }
  wait_async0();
  __syncthreads();
  const int row0 = (tid >> 5) * 16;
  v16bf a[4];
#pragma unroll
  for (int k = 0; k < 4; ++k) a[k] = frag_a(sX, row0, k * 32, 128);
#pragma unroll
  for (int n = 0; n < 4; ++n) {
    v8f ck = {}; v8f cq = {};
#pragma unroll
    for (int k = 0; k < 4; ++k) {
      ck = __builtin_amdgcn_wmma_f32_16x16x32_bf16(false, a[k], false,
             frag_b(sK, k * 32, n * 16, 128), (short)0, ck, false, false);
      cq = __builtin_amdgcn_wmma_f32_16x16x32_bf16(false, a[k], false,
             frag_b(sQ, k * 32, n * 16, 128), (short)0, cq, false, false);
    }
    store_c(Kb + r0 * 64, 64, row0, n * 16, ck, sbk);
    store_c(Qb + r0 * 64, 64, row0, n * 16, cq, sbq);
  }
}

// ---- Kernel 2: binding sign test + per-row prefix scan -> seg, chunk starts
__global__ __launch_bounds__(256) void k_scan(const float* __restrict__ Kb,
    const float* __restrict__ Qb, int* __restrict__ seg,
    int* __restrict__ spos, int* __restrict__ nchk) {
  __shared__ int sf[256];
  __shared__ int carry;
  const int b = blockIdx.x;
  const int tid = threadIdx.x;
  if (tid == 0) carry = 0;
  __syncthreads();
  for (int base = 0; base < T_; base += 256) {
    const int t = base + tid;
    int start = 1;
    if (t > 0) {
      const float4* kr = (const float4*)(Kb + ((long)b * T_ + t - 1) * 64);
      const float4* qr = (const float4*)(Qb + ((long)b * T_ + t) * 64);
      float s = 0.f;
#pragma unroll
      for (int i = 0; i < 16; ++i) {
        const float4 kv = kr[i], qv = qr[i];
        s += kv.x * qv.x + kv.y * qv.y + kv.z * qv.z + kv.w * qv.w;
      }
      start = (s <= 0.f) ? 1 : 0;  // sigmoid(s/8) > 0.5  <=>  s > 0
    }
    sf[tid] = start;
    __syncthreads();
    for (int off = 1; off < 256; off <<= 1) {
      int v = 0;
      if (tid >= off) v = sf[tid - off];
      __syncthreads();
      sf[tid] += v;
      __syncthreads();
    }
    const int sid = carry + sf[tid] - 1;
    seg[(long)b * T_ + t] = sid;
    if (start) spos[b * (T_ + 1) + sid] = t;
    __syncthreads();
    if (tid == 0) carry += sf[255];
    __syncthreads();
  }
  if (tid == 0) { nchk[b] = carry; spos[b * (T_ + 1) + carry] = T_; }
}

// ---- Kernel 3: contiguous-run segment mean -> bf16 chunk_mean --------------
__global__ __launch_bounds__(256) void k_means(const float* __restrict__ x,
    const int* __restrict__ spos, const int* __restrict__ nchk,
    unsigned short* __restrict__ cmh) {
  const int slot = blockIdx.x * 2 + (threadIdx.x >> 7);
  const int d = threadIdx.x & 127;
  const int b = slot >> 14;          // T_ = 16384
  const int c = slot & (T_ - 1);
  float m = 0.f;
  if (c < nchk[b]) {
    const int s = spos[b * (T_ + 1) + c];
    const int e = spos[b * (T_ + 1) + c + 1];
    float acc = 0.f;
    for (int t = s; t < e; ++t) acc += x[((long)b * T_ + t) * 128 + d];
    m = acc / (float)(e - s);
  }
  cmh[((long)b * T_ + c) * 128 + d] = f32_bf16(m);
}

// ---- Kernel 4: fused  Wa1 + LayerNorm + ReLU + Wcomb  (bf16 WMMA) ----------
__global__ __launch_bounds__(128) void k_mlp(const unsigned short* __restrict__ cmh,
    const unsigned short* __restrict__ uW1T, const unsigned short* __restrict__ uWcT,
    const float* __restrict__ ba1, const float* __restrict__ g,
    const float* __restrict__ be,  const float* __restrict__ bc,
    float* __restrict__ infl) {
  extern __shared__ char smem[];
  float* fH = (float*)smem;                           // 64x128 fp32 (LN input)  32KB
  float* p1 = fH + 64 * 128;                          // ba1
  float* pg = p1 + 128;                               // ln gamma
  float* pb = pg + 128;                               // ln beta
  float* pc = pb + 128;                               // bcomb                    2KB
  unsigned short* uW1 = (unsigned short*)(pc + 128);  // 128x128 bf16 col-major  32KB
  unsigned short* uWc = uW1 + 128 * 128;              // 128x128 bf16 col-major  32KB
  unsigned short* uIn = uWc + 128 * 128;              // 64x128 bf16 row-major   16KB
  unsigned short* uR  = uIn + 64 * 128;               // 64x128 bf16 row-major   16KB
  const int tid = threadIdx.x;
  const long r0 = (long)blockIdx.x * 64;
  stage_async(uW1, uW1T, 128 * 128 * 2);              // DMA weight + input tiles
  stage_async(uWc, uWcT, 128 * 128 * 2);
  stage_async(uIn, cmh + r0 * 128, 64 * 128 * 2);
  if (tid < 128) { p1[tid] = ba1[tid]; pg[tid] = g[tid]; pb[tid] = be[tid]; pc[tid] = bc[tid]; }
  wait_async0();
  __syncthreads();
  const int row0 = (tid >> 5) * 16;
  {                                                   // H1 = cm @ Wa1 + ba1
    v16bf a[4];
#pragma unroll
    for (int k = 0; k < 4; ++k) a[k] = frag_a(uIn, row0, k * 32, 128);
#pragma unroll
    for (int n = 0; n < 8; ++n) {
      v8f acc = {};
#pragma unroll
      for (int k = 0; k < 4; ++k)
        acc = __builtin_amdgcn_wmma_f32_16x16x32_bf16(false, a[k], false,
                frag_b(uW1, k * 32, n * 16, 128), (short)0, acc, false, false);
      store_c(fH, 128, row0, n * 16, acc, p1);
    }
  }
  __syncthreads();
  if (tid < 64) {                                     // LayerNorm (biased var) + ReLU
    const float* hr = fH + tid * 128;
    float s = 0.f, s2 = 0.f;
    for (int d = 0; d < 128; ++d) { const float v = hr[d]; s += v; s2 += v * v; }
    const float mu  = s * (1.f / 128.f);
    const float var = s2 * (1.f / 128.f) - mu * mu;
    const float rs  = rsqrtf(var + 1e-5f);
    for (int d = 0; d < 128; ++d) {
      const float v = (hr[d] - mu) * rs * pg[d] + pb[d];
      uR[tid * 128 + d] = f32_bf16(v > 0.f ? v : 0.f);
    }
  }
  __syncthreads();
  {                                                   // infl = relu @ Wcomb + bcomb
    v16bf a[4];
#pragma unroll
    for (int k = 0; k < 4; ++k) a[k] = frag_a(uR, row0, k * 32, 128);
    float* dst = infl + r0 * 128;
#pragma unroll
    for (int n = 0; n < 8; ++n) {
      v8f acc = {};
#pragma unroll
      for (int k = 0; k < 4; ++k)
        acc = __builtin_amdgcn_wmma_f32_16x16x32_bf16(false, a[k], false,
                frag_b(uWc, k * 32, n * 16, 128), (short)0, acc, false, false);
      store_c(dst, 128, row0, n * 16, acc, pc);
    }
  }
}

// ---- Kernel 5: out = x * sigmoid(infl[seg]) + x  (float4) ------------------
__global__ __launch_bounds__(256) void k_gate(const float* __restrict__ x,
    const float* __restrict__ infl, const int* __restrict__ seg,
    float* __restrict__ out) {
  const long i = (long)blockIdx.x * 256 + threadIdx.x;   // BT*32 float4 units
  const long token = i >> 5;
  const int  d4 = (int)(i & 31);
  const int  b  = (int)(token >> 14);
  const int  sg = seg[token];
  const float4 xv = ((const float4*)x)[i];
  const float4 iv = ((const float4*)infl)[(((long)b * T_) + sg) * 32 + d4];
  float4 o;
  o.x = xv.x + xv.x / (1.f + __expf(-iv.x));
  o.y = xv.y + xv.y / (1.f + __expf(-iv.y));
  o.z = xv.z + xv.z / (1.f + __expf(-iv.z));
  o.w = xv.w + xv.w / (1.f + __expf(-iv.w));
  ((float4*)out)[i] = o;
}

extern "C" void kernel_launch(void* const* d_in, const int* in_sizes, int n_in,
                              void* d_out, int out_size, void* d_ws, size_t ws_size,
                              hipStream_t stream) {
  (void)in_sizes; (void)n_in; (void)out_size; (void)ws_size;
  const float* x   = (const float*)d_in[0];
  const float* Wk  = (const float*)d_in[1];
  const float* bk  = (const float*)d_in[2];
  const float* Wq  = (const float*)d_in[3];
  const float* bq  = (const float*)d_in[4];
  const float* Wa1 = (const float*)d_in[5];
  const float* ba1 = (const float*)d_in[6];
  const float* lng = (const float*)d_in[7];
  const float* lnb = (const float*)d_in[8];
  const float* Wa2 = (const float*)d_in[9];
  const float* ba2 = (const float*)d_in[10];
  const float* Wd  = (const float*)d_in[11];
  const float* bd  = (const float*)d_in[12];
  float* out = (float*)d_out;

  // Workspace layout (~135.5 MB), all sections 16B aligned:
  //   [0, BT*64)        Kbuf   \__ dead after k_scan; region reused as infl (BT*128 f32)
  //   [BT*64, BT*128)   Qbuf   /
  //   [BT*128, ...)     chunk_mean (BT*128 bf16)
  //   then: seg (BT i32), spos (B*(T+1) i32), nchk (B i32), bcomb (128 f32),
  //         WkT/WqT (64x128 bf16 each), Wa1T/WcombT (128x128 bf16 each)
  float* base = (float*)d_ws;
  float* Kb   = base;
  float* Qb   = base + (size_t)BT_ * 64;
  float* infl = base;                          // overlays K/Q once scan is done
  unsigned short* cmh = (unsigned short*)(base + (size_t)BT_ * 128);
  int*   seg  = (int*)(base + (size_t)2 * BT_ * 128);
  int*   spos = seg + BT_;
  int*   nchk = spos + B_ * (T_ + 1);
  float* bc   = (float*)(nchk + B_ + 56);      // pad to 16B boundary
  unsigned short* uWkT = (unsigned short*)(bc + 128);
  unsigned short* uWqT = uWkT + 64 * 128;
  unsigned short* uW1T = uWqT + 64 * 128;
  unsigned short* uWcT = uW1T + 128 * 128;

  k_prep <<<(128 * 128) / 256, 256, 0, stream>>>(Wk, Wq, Wa1, Wa2, ba2, Wd, bd,
                                                 uWkT, uWqT, uW1T, uWcT, bc);
  k_proj <<<BT_ / 64, 128, 0, stream>>>(x, uWkT, uWqT, bk, bq, Kb, Qb);
  k_scan <<<B_, 256, 0, stream>>>(Kb, Qb, seg, spos, nchk);
  k_means<<<BT_ / 2, 256, 0, stream>>>(x, spos, nchk, cmh);
  const size_t mlp_lds = (size_t)(64 * 128 + 4 * 128) * sizeof(float)
                       + (size_t)(2 * 128 * 128 + 2 * 64 * 128) * sizeof(unsigned short);
  k_mlp  <<<BT_ / 64, 128, mlp_lds, stream>>>(cmh, uW1T, uWcT, ba1, lng, lnb, bc, infl);
  k_gate <<<(BT_ * 32) / 256, 256, 0, stream>>>(x, infl, seg, out);
}
